// DefaultMapping_1563368095943
// MI455X (gfx1250) — compile-verified, matrix operand order checked by reading, not run on hardware
//
#include <hip/hip_runtime.h>
#include <hip/hip_bf16.h>

#define B_  16
#define L_  4096
#define RP_ 8192
#define R_  512
#define P_  2048
#define D_  128

typedef __attribute__((ext_vector_type(16))) _Float16 v16h;
typedef __attribute__((ext_vector_type(8)))  _Float16 v8h;
typedef __attribute__((ext_vector_type(8)))  float    v8f;

union FragH {
    v16h v;
    v8h  h[2];
};

// ---------------------------------------------------------------------------
// Kernel 1: zero counts + cursors (contiguous ints)
// ---------------------------------------------------------------------------
__global__ void zero_int_kernel(int* __restrict__ p, int n) {
    int i = blockIdx.x * 256 + threadIdx.x;
    if (i < n) p[i] = 0;
}

// ---------------------------------------------------------------------------
// Kernel 2: histogram road_idx per batch (int atomics, 131K total)
// ---------------------------------------------------------------------------
__global__ void count_kernel(const int* __restrict__ road_idx,
                             int* __restrict__ counts) {
    int j = blockIdx.x * 256 + threadIdx.x;     // < B*RP
    if (j < B_ * RP_) {
        int b = j / RP_;
        atomicAdd(&counts[b * R_ + road_idx[j]], 1);
    }
}

// ---------------------------------------------------------------------------
// Kernel 3: per-batch exclusive scan over R=512 buckets (tiny, serial)
// ---------------------------------------------------------------------------
__global__ void scan_kernel(const int* __restrict__ counts,
                            int* __restrict__ offsets) {
    int b = blockIdx.x;                         // one block per batch
    if (threadIdx.x == 0) {
        int acc = 0;
        for (int r = 0; r < R_; ++r) {
            offsets[b * R_ + r] = acc;
            acc += counts[b * R_ + r];
        }
    }
}

// ---------------------------------------------------------------------------
// Kernel 4: scatter road-row ids into bucket-ordered list
// ---------------------------------------------------------------------------
__global__ void scatter_kernel(const int* __restrict__ road_idx,
                               const int* __restrict__ offsets,
                               int* __restrict__ cursors,
                               int* __restrict__ order) {
    int j = blockIdx.x * 256 + threadIdx.x;     // < B*RP
    if (j < B_ * RP_) {
        int b  = j / RP_;
        int rj = j % RP_;
        int r  = road_idx[j];
        int slot = atomicAdd(&cursors[b * R_ + r], 1);
        order[(size_t)b * RP_ + offsets[b * R_ + r] + slot] = rj;
    }
}

// ---------------------------------------------------------------------------
// Kernel 5: gather-pool. One block per (b,r); 128 threads stream the ~16
// member rows coalesced along D, mean, emit packed f16 B-operand.
// Road features are read exactly once, no float atomics.
// ---------------------------------------------------------------------------
__global__ void pool_gather_kernel(const float* __restrict__ feat,
                                   const int*   __restrict__ counts,
                                   const int*   __restrict__ offsets,
                                   const int*   __restrict__ order,
                                   _Float16* __restrict__ pooledh) {
    int br = blockIdx.x;                        // b*R + r
    int b  = br / R_;
    int d  = threadIdx.x;                       // 0..127
    int cnt = counts[br];
    int off = offsets[br];
    const int*   ord   = order + (size_t)b * RP_ + off;
    const float* roadB = feat + ((size_t)b * (L_ + RP_) + L_) * D_;
    float s = 0.0f;
    for (int i = 0; i < cnt; ++i)
        s += roadB[(size_t)ord[i] * D_ + d];
    float m = s / fmaxf((float)cnt, 1.0f);
    pooledh[(size_t)br * D_ + d] = (_Float16)m;
}

// ---------------------------------------------------------------------------
// Kernel 6: fused  sim = lane @ pooled^T / sqrt(D)  +  row softmax.
// One block = one (batch, 16-row lane tile). 256 threads = 8 wave32.
// Wave w computes N in [w*64, w*64+64) via 4x(K-steps) x 4x(N-tiles) of
// v_wmma_f32_16x16x32_f16.  Epilogue: cross-wave softmax in LDS, coalesced
// float4 stores of both sim and sim_softmax.
// ---------------------------------------------------------------------------
__global__ __launch_bounds__(256) void sim_softmax_kernel(
        const float*    __restrict__ feat,
        const _Float16* __restrict__ pooledh,
        float* __restrict__ sim_out,
        float* __restrict__ smax_out) {
    __shared__ __align__(16) _Float16 laneH[16 * 128];   // 4 KB  A-tile (f16)
    __shared__ float simS[16 * 512];                     // 32 KB sim tile
    __shared__ float red[16 * 16];                       // 1 KB  reduction
    __shared__ float rowmax[16];
    __shared__ float rowsum[16];

    const int blk  = blockIdx.x;
    const int b    = blk / (L_ / 16);
    const int lt   = blk % (L_ / 16);
    const int row0 = lt * 16;
    const int tid  = threadIdx.x;

    // ---- stage 16x128 lane tile into LDS as f16 (rows are contiguous) ----
    const float* laneG = feat + ((size_t)b * (L_ + RP_) + row0) * D_;
    #pragma unroll
    for (int it = 0; it < 2; ++it) {
        int g = tid + it * 256;                 // 512 float4 groups total
        float4 f = ((const float4*)laneG)[g];
        laneH[g * 4 + 0] = (_Float16)f.x;
        laneH[g * 4 + 1] = (_Float16)f.y;
        laneH[g * 4 + 2] = (_Float16)f.z;
        laneH[g * 4 + 3] = (_Float16)f.w;
    }
    __syncthreads();

    const int wv   = tid >> 5;
    const int lane = tid & 31;
    const int m16  = lane & 15;            // row (A) / col (B) within tile
    const int kb   = (lane >> 4) << 3;     // K sub-offset: 0 | 8

    v8f acc0 = {}; v8f acc1 = {}; v8f acc2 = {}; v8f acc3 = {};

    const _Float16* poolB = pooledh + (size_t)b * R_ * D_;

    #pragma unroll
    for (int kc = 0; kc < 4; ++kc) {       // D=128 in K=32 chunks
        const int kof = kc * 32 + kb;

        FragH a;                            // A: lane = M, ISA 16-bit layout
        a.h[0] = *(const v8h*)&laneH[m16 * 128 + kof];
        a.h[1] = *(const v8h*)&laneH[m16 * 128 + kof + 16];

        #pragma unroll
        for (int nt = 0; nt < 4; ++nt) {   // 4 N-tiles of 16 per wave
            const int n = wv * 64 + nt * 16 + m16;
            const _Float16* bp = poolB + (size_t)n * D_ + kof;
            FragH bf;                       // B: lane = N, same K striping
            bf.h[0] = *(const v8h*)bp;
            bf.h[1] = *(const v8h*)(bp + 16);
            v8f c = (nt == 0) ? acc0 : (nt == 1) ? acc1 : (nt == 2) ? acc2 : acc3;
            c = __builtin_amdgcn_wmma_f32_16x16x32_f16(
                    false, a.v, false, bf.v, (short)0, c, false, false);
            if      (nt == 0) acc0 = c;
            else if (nt == 1) acc1 = c;
            else if (nt == 2) acc2 = c;
            else              acc3 = c;
        }
    }

    // ---- dump accumulators (scaled by 1/sqrt(D)) into LDS sim tile ----
    const float scale = 0.08838834764831845f;   // 1/sqrt(128)
    const int mh = (lane >> 4) << 3;            // row offset 0 | 8
    #pragma unroll
    for (int nt = 0; nt < 4; ++nt) {
        const int ncol = wv * 64 + nt * 16 + m16;
        v8f c = (nt == 0) ? acc0 : (nt == 1) ? acc1 : (nt == 2) ? acc2 : acc3;
        #pragma unroll
        for (int v = 0; v < 8; ++v)
            simS[(mh + v) * 512 + ncol] = c[v] * scale;
    }
    __syncthreads();

    // ---- softmax across R=512 per row: 16 rows x 16 segments of 32 ----
    const int row = tid >> 4;
    const int seg = tid & 15;
    const float* sr = &simS[row * 512 + seg * 32];

    float mx = -1e30f;
    #pragma unroll
    for (int i = 0; i < 32; ++i) mx = fmaxf(mx, sr[i]);
    red[row * 16 + seg] = mx;
    __syncthreads();
    if (seg == 0) {
        float m = red[row * 16];
        #pragma unroll
        for (int i = 1; i < 16; ++i) m = fmaxf(m, red[row * 16 + i]);
        rowmax[row] = m;
    }
    __syncthreads();
    const float rm = rowmax[row];
    float s = 0.0f;
    #pragma unroll
    for (int i = 0; i < 32; ++i) s += __expf(sr[i] - rm);
    red[row * 16 + seg] = s;
    __syncthreads();
    if (seg == 0) {
        float t = 0.0f;
        #pragma unroll
        for (int i = 0; i < 16; ++i) t += red[row * 16 + i];
        rowsum[row] = t;
    }
    __syncthreads();
    const float inv = 1.0f / rowsum[row];

    // ---- coalesced float4 writes of sim and softmax(sim) ----
    const size_t obase = ((size_t)b * L_ + row0 + row) * R_ + seg * 32;
    float4* so = (float4*)(sim_out + obase);
    float4* xo = (float4*)(smax_out + obase);
    #pragma unroll
    for (int i = 0; i < 8; ++i) {
        float4 v, e;
        v.x = sr[i * 4 + 0]; v.y = sr[i * 4 + 1];
        v.z = sr[i * 4 + 2]; v.w = sr[i * 4 + 3];
        e.x = __expf(v.x - rm) * inv; e.y = __expf(v.y - rm) * inv;
        e.z = __expf(v.z - rm) * inv; e.w = __expf(v.w - rm) * inv;
        so[i] = v;
        xo[i] = e;
    }
}

// ---------------------------------------------------------------------------
// Kernel 7: path_sim gather (L2-hot rows of sim) + path_gt
// grid = B*P blocks, 128 threads (one float4 each -> 512 floats/row)
// ---------------------------------------------------------------------------
__global__ void path_kernel(const float* __restrict__ sim,
                            const int*   __restrict__ path_inverse,
                            const int*   __restrict__ gt,
                            float* __restrict__ path_sim,
                            float* __restrict__ path_gt) {
    int bp = blockIdx.x;            // 0 .. B*P-1
    int b  = bp / P_;
    int l  = path_inverse[bp];
    const float4* src = (const float4*)(sim + ((size_t)b * L_ + l) * R_);
    float4*       dst = (float4*)(path_sim + (size_t)bp * R_);
    dst[threadIdx.x] = src[threadIdx.x];
    if (threadIdx.x == 0) path_gt[bp] = (float)gt[b * L_ + l];
}

// ---------------------------------------------------------------------------
extern "C" void kernel_launch(void* const* d_in, const int* in_sizes, int n_in,
                              void* d_out, int out_size, void* d_ws, size_t ws_size,
                              hipStream_t stream) {
    (void)in_sizes; (void)n_in; (void)out_size; (void)ws_size;

    const float* feat         = (const float*)d_in[0];
    const int*   road_idx     = (const int*)d_in[1];
    const int*   path_inverse = (const int*)d_in[2];
    const int*   gt           = (const int*)d_in[3];

    float* out  = (float*)d_out;
    float* sim  = out;                                   // B*L*R
    float* smax = sim  + (size_t)B_ * L_ * R_;           // B*L*R
    float* psim = smax + (size_t)B_ * L_ * R_;           // B*P*R
    float* pgt  = psim + (size_t)B_ * P_ * R_;           // B*P

    // workspace layout (ints first so counts+cursors zero contiguously):
    //   counts  (B*R)  | cursors (B*R) | offsets (B*R) | order (B*RP)
    //   pooledh (B*R*D f16)
    int* counts  = (int*)d_ws;
    int* cursors = counts  + B_ * R_;
    int* offsets = cursors + B_ * R_;
    int* order   = offsets + B_ * R_;
    _Float16* pooledh = (_Float16*)(order + (size_t)B_ * RP_);

    const int nz = 2 * B_ * R_;                      // counts + cursors
    zero_int_kernel<<<(nz + 255) / 256, 256, 0, stream>>>(counts, nz);

    count_kernel<<<(B_ * RP_ + 255) / 256, 256, 0, stream>>>(road_idx, counts);

    scan_kernel<<<B_, 32, 0, stream>>>(counts, offsets);

    scatter_kernel<<<(B_ * RP_ + 255) / 256, 256, 0, stream>>>(
        road_idx, offsets, cursors, order);

    pool_gather_kernel<<<B_ * R_, 128, 0, stream>>>(
        feat, counts, offsets, order, pooledh);

    sim_softmax_kernel<<<B_ * (L_ / 16), 256, 0, stream>>>(feat, pooledh, sim, smax);

    path_kernel<<<B_ * P_, 128, 0, stream>>>(sim, path_inverse, gt, psim, pgt);
}
